// RNN_Ecoder_85959475462790
// MI455X (gfx1250) — matrix operand order repaired and checked by reading, not compiled, not run on hardware
//
#include <hip/hip_runtime.h>

// MI455X / gfx1250, wave32. WMMA f32 <- f16 16x16x32.
typedef __attribute__((ext_vector_type(16))) _Float16 v16h;
typedef __attribute__((ext_vector_type(8)))  float    v8f;
typedef __attribute__((ext_vector_type(4)))  int      v4i;

#define T_STEPS 256
#define BATCH   256
#define IDIM    512
#define HDIM    512
#define KTILES  (HDIM / 32)
#define LDS_STRIDE 520   // 512 + 8 halves (16B pad) -> conflict-free ds_load_b128
#define XP_STRIDE  516   // 512 + 4 floats: staggers banks between lane halves

#if __has_builtin(__builtin_amdgcn_global_load_async_to_lds_b128)
#define HAVE_ASYNC_LDS 1
typedef __attribute__((address_space(1))) v4i gv4i;   // global int4
typedef __attribute__((address_space(3))) v4i lv4i;   // LDS int4
#else
#define HAVE_ASYNC_LDS 0
#endif

union AFrag { v16h v; _Float16 h[16]; uint4 q[2]; };

// B fragment: lane holds col (L%16) of the 32x16 tile = row of W (NxK),
// K = kc + (L/16)*8 .. +8 and kc + 16 + (L/16)*8 .. +8 -> two 16B chunks.
__device__ __forceinline__ void load_w_frag(AFrag& b, const _Float16* __restrict__ W,
                                            int nrow, int kc, int half) {
    const _Float16* bp = W + (size_t)nrow * HDIM + kc + half * 8;
    b.q[0] = *(const uint4*)(bp);
    b.q[1] = *(const uint4*)(bp + 16);
}

__device__ __forceinline__ void wait_asynccnt0() {
#if __has_builtin(__builtin_amdgcn_s_wait_asynccnt)
    __builtin_amdgcn_s_wait_asynccnt(0);
#else
    asm volatile("s_wait_asynccnt 0x0" ::: "memory");
#endif
}

// ---------------------------------------------------------------------------
// f32 -> f16 conversion (weights only)
// ---------------------------------------------------------------------------
__global__ void cvt_f32_to_f16(const float* __restrict__ src,
                               _Float16* __restrict__ dst, int n) {
    int i = blockIdx.x * blockDim.x + threadIdx.x;
    if (i < n) dst[i] = (_Float16)src[i];
}

// ---------------------------------------------------------------------------
// Phase 1: x_proj[m,n] = sum_k inputs[m,k]*W_ih[n,k] + b_ih[n] + b_hh[n]
// M=65536, N=512, K=512. 8 waves/WG stacked in M; wave tile 16x64.
// Register double-buffered: K-chunk k+1 loads overlap chunk k's WMMAs.
// ---------------------------------------------------------------------------
__global__ __launch_bounds__(256)
void xproj_gemm(const float* __restrict__ X,        // [M][K] f32
                const _Float16* __restrict__ Wih,   // [N][K] f16
                const float* __restrict__ b_ih,
                const float* __restrict__ b_hh,
                float* __restrict__ XP)             // [M][N] f32
{
    const int lane = threadIdx.x & 31;
    const int wave = threadIdx.x >> 5;
    const int r    = lane & 15;
    const int half = lane >> 4;
    const int nBlock = blockIdx.x & 7;
    const int mBlock = blockIdx.x >> 3;
    const int m0 = mBlock * 128 + wave * 16;
    const int n0 = nBlock * 64;

    v8f acc[4];
    #pragma unroll
    for (int j = 0; j < 4; ++j) {
        float bias = b_ih[n0 + j*16 + r] + b_hh[n0 + j*16 + r];
        #pragma unroll
        for (int e = 0; e < 8; ++e) acc[j][e] = bias;
    }

    const float* arow = X + (size_t)(m0 + r) * IDIM;
    auto load_a = [&](AFrag& a, int kc) {
        const float* p0 = arow + kc + half * 8;
        #pragma unroll
        for (int e = 0; e < 8; ++e) a.h[e]     = (_Float16)p0[e];
        #pragma unroll
        for (int e = 0; e < 8; ++e) a.h[8 + e] = (_Float16)p0[16 + e];
    };

    AFrag a[2], bf[2][4];
    load_a(a[0], 0);
    #pragma unroll
    for (int j = 0; j < 4; ++j) load_w_frag(bf[0][j], Wih, n0 + j*16 + r, 0, half);

    #pragma unroll
    for (int kci = 0; kci < KTILES; ++kci) {
        const int cb = kci & 1, nb = cb ^ 1;
        if (kci + 1 < KTILES) {
            const int kn = (kci + 1) * 32;
            load_a(a[nb], kn);
            #pragma unroll
            for (int j = 0; j < 4; ++j)
                load_w_frag(bf[nb][j], Wih, n0 + j*16 + r, kn, half);
        }
        #pragma unroll
        for (int j = 0; j < 4; ++j)
            acc[j] = __builtin_amdgcn_wmma_f32_16x16x32_f16(
                false, a[cb].v, false, bf[cb][j].v, (short)0, acc[j], false, false);
    }

    #pragma unroll
    for (int j = 0; j < 4; ++j) {
        int col = n0 + j*16 + r;
        #pragma unroll
        for (int e = 0; e < 8; ++e)
            XP[(size_t)(m0 + e + half*8) * HDIM + col] = acc[j][e];
    }
}

// ---------------------------------------------------------------------------
// Phase 2: sequential scan, batch-partitioned (no cross-WG sync).
// 16 WGs x 16 batch rows; h double-buffered in LDS; W_hh streamed from L2.
// x_proj tile for step t+1 async-copied to LDS during step t's GEMM.
// ---------------------------------------------------------------------------
__global__ __launch_bounds__(256)
void rnn_scan(const float* __restrict__ XP,        // [T][B][H] f32
              const _Float16* __restrict__ Whh,    // [H(N)][H(K)] f16
              float* __restrict__ Hout)            // [B][H] f32
{
    __shared__ __align__(32) _Float16 hbuf[2][16][LDS_STRIDE];
#if HAVE_ASYNC_LDS
    __shared__ __align__(16) float xpbuf[2][16][XP_STRIDE];
#endif
    const int tid  = threadIdx.x;
    const int lane = tid & 31;
    const int wave = tid >> 5;
    const int r    = lane & 15;
    const int half = lane >> 4;
    const int bBase = blockIdx.x * 16;
    const int n0    = wave * 64;

    for (int i = tid; i < 2 * 16 * LDS_STRIDE; i += 256)
        ((_Float16*)hbuf)[i] = (_Float16)0.f;

#if HAVE_ASYNC_LDS
    {   // pre-stage x_proj tile for t = 0 (16 rows x 2KB = 32KB, 16B/lane chunks)
        const float* src0 = XP + (size_t)bBase * HDIM;
        #pragma unroll
        for (int issue = 0; issue < 8; ++issue) {
            int idx = issue * 256 + tid;
            int row = idx >> 7, c = (idx & 127) * 4;
            __builtin_amdgcn_global_load_async_to_lds_b128(
                (gv4i*)(src0 + (size_t)row * HDIM + c),
                (lv4i*)(&xpbuf[0][row][c]), 0, 0);
        }
        wait_asynccnt0();
    }
#endif
    __syncthreads();

    int cur = 0;
    for (int t = 0; t < T_STEPS; ++t) {
        const int nxt = cur ^ 1;

#if HAVE_ASYNC_LDS
        if (t + 1 < T_STEPS) {   // overlap next tile's copy with this step's GEMM
            const float* srcn = XP + ((size_t)(t + 1) * BATCH + bBase) * HDIM;
            #pragma unroll
            for (int issue = 0; issue < 8; ++issue) {
                int idx = issue * 256 + tid;
                int row = idx >> 7, c = (idx & 127) * 4;
                __builtin_amdgcn_global_load_async_to_lds_b128(
                    (gv4i*)(srcn + (size_t)row * HDIM + c),
                    (lv4i*)(&xpbuf[nxt][row][c]), 0, 0);
            }
        }
#endif
        // C init = x_proj[t, bBase:bBase+16, :]
        v8f acc[4];
#if HAVE_ASYNC_LDS
        #pragma unroll
        for (int j = 0; j < 4; ++j) {
            int col = n0 + j*16 + r;
            #pragma unroll
            for (int e = 0; e < 8; ++e)
                acc[j][e] = xpbuf[cur][e + half*8][col];
        }
#else
        const float* xp = XP + ((size_t)t * BATCH + bBase) * HDIM;
        if (t + 1 < T_STEPS)
            __builtin_prefetch(xp + (size_t)BATCH * HDIM + tid * 32, 0, 1);
        #pragma unroll
        for (int j = 0; j < 4; ++j) {
            int col = n0 + j*16 + r;
            #pragma unroll
            for (int e = 0; e < 8; ++e)
                acc[j][e] = xp[(size_t)(e + half*8) * HDIM + col];
        }
#endif
        // h @ W_hh^T over K=512, register double-buffered
        const _Float16* hrow = &hbuf[cur][r][0];
        AFrag a[2], bf[2][4];
        {
            const _Float16* ap = hrow + half * 8;
            a[0].q[0] = *(const uint4*)(ap);
            a[0].q[1] = *(const uint4*)(ap + 16);
            #pragma unroll
            for (int j = 0; j < 4; ++j)
                load_w_frag(bf[0][j], Whh, n0 + j*16 + r, 0, half);
        }
        #pragma unroll
        for (int kci = 0; kci < KTILES; ++kci) {
            const int cb = kci & 1, nb = cb ^ 1;
            if (kci + 1 < KTILES) {
                const int kn = (kci + 1) * 32;
                const _Float16* ap = hrow + kn + half * 8;
                a[nb].q[0] = *(const uint4*)(ap);
                a[nb].q[1] = *(const uint4*)(ap + 16);
                #pragma unroll
                for (int j = 0; j < 4; ++j)
                    load_w_frag(bf[nb][j], Whh, n0 + j*16 + r, kn, half);
            }
            #pragma unroll
            for (int j = 0; j < 4; ++j)
                acc[j] = __builtin_amdgcn_wmma_f32_16x16x32_f16(
                    false, a[cb].v, false, bf[cb][j].v, (short)0, acc[j], false, false);
        }

        // tanh; publish h (f16 LDS) + f32 output (last step wins)
        #pragma unroll
        for (int j = 0; j < 4; ++j) {
            int col = n0 + j*16 + r;
            #pragma unroll
            for (int e = 0; e < 8; ++e) {
                int row = e + half * 8;
                float v = tanhf(acc[j][e]);
                Hout[(size_t)(bBase + row) * HDIM + col] = v;
                hbuf[nxt][row][col] = (_Float16)v;
            }
        }
#if HAVE_ASYNC_LDS
        wait_asynccnt0();
#endif
        __syncthreads();
        cur = nxt;
    }
}

// ---------------------------------------------------------------------------
extern "C" void kernel_launch(void* const* d_in, const int* in_sizes, int n_in,
                              void* d_out, int out_size, void* d_ws, size_t ws_size,
                              hipStream_t stream) {
    const float* inputs = (const float*)d_in[0];   // [T,B,I]
    const float* W_ih   = (const float*)d_in[1];   // [H,I]
    const float* W_hh   = (const float*)d_in[2];   // [H,H]
    const float* b_ih   = (const float*)d_in[3];   // [H]
    const float* b_hh   = (const float*)d_in[4];   // [H]
    float* out = (float*)d_out;                    // [B,H]

    // ws layout: [x_proj f32 128MB][W_ih f16 512KB][W_hh f16 512KB]
    char* ws = (char*)d_ws;
    float* xp = (float*)ws;
    size_t off = (size_t)T_STEPS * BATCH * HDIM * sizeof(float);
    _Float16* Wih16 = (_Float16*)(ws + off);
    _Float16* Whh16 = (_Float16*)(ws + off + (size_t)HDIM * IDIM * sizeof(_Float16));

    const int nW = HDIM * IDIM;
    cvt_f32_to_f16<<<(nW + 255) / 256, 256, 0, stream>>>(W_ih, Wih16, nW);
    cvt_f32_to_f16<<<(nW + 255) / 256, 256, 0, stream>>>(W_hh, Whh16, nW);

    const int grid1 = (T_STEPS * BATCH / 128) * (HDIM / 64);   // 4096
    xproj_gemm<<<grid1, 256, 0, stream>>>(inputs, Wih16, b_ih, b_hh, xp);

    rnn_scan<<<BATCH / 16, 256, 0, stream>>>(xp, Whh16, out);
}
